// DecoderWithAttention_12618613915848
// MI455X (gfx1250) — matrix operand order repaired and checked by reading, not compiled, not run on hardware
//
#include <hip/hip_runtime.h>
#include <hip/hip_bf16.h>

// ---------------------------------------------------------------------------
// Show-Attend-Tell decoder for MI455X (gfx1250, wave32).
// All GEMMs via v_wmma_f32_16x16x32_bf16 (bf16 in, f32 accumulate).
// Weights converted f32->bf16 and pre-swizzled into WMMA fragment layout once;
// activations (h, x=[emb|gated awe], mean_enc, enc) packed per use.
// Big GEMMs use 2x2 output-tile register blocking: 1 b128 load per wmma
// instead of 2 (2x arithmetic intensity at the VMEM issue port).
// enc (103MB) + att1 (26MB) + weights stay resident in the 192MB L2 across
// the 51 sequential LSTM steps.
// ---------------------------------------------------------------------------

#define Ec   2048   // encoder dim E
#define Dc   512    // decoder dim D
#define Ac   512    // attention dim A
#define EMBc 512    // embedding dim
#define Vc   10000  // vocab
#define Bc   64     // batch
#define Pc   196    // pixels (14*14)
#define Lc   52     // max caption len
#define Tc   51     // decode steps (L-1)
#define KXc  (EMBc + Ec + Dc)   // 3072: fused LSTM input [emb | awe | h]

typedef __attribute__((ext_vector_type(16))) __bf16 v16bf;
typedef __attribute__((ext_vector_type(8)))  float  v8f;

// float -> bf16, round-to-nearest-even
__device__ __forceinline__ unsigned short f2bf(float f) {
    unsigned int u = __float_as_uint(f);
    u += 0x7FFFu + ((u >> 16) & 1u);
    return (unsigned short)(u >> 16);
}
__device__ __forceinline__ float sigmoidf_(float x) {
    return 1.0f / (1.0f + __expf(-x));
}

// --- WMMA fragment index helpers (CDNA5 ISA 7.12.2 layouts, wave32) --------
// A matrix (16x32 bf16): lane l<16 holds M=l, halves 0..7 => K=0..7,
// halves 8..15 => K=16..23; lane l>=16 holds M=l-16, K=8..15 / K=24..31.
__device__ __forceinline__ size_t a_frag_idx(int m, int k, int Kt) {
    int mt = m >> 4, kt = k >> 5;
    int mo = m & 15, ko = k & 31;
    int lane = mo + (((ko >> 3) & 1) << 4);
    int h    = (ko & 7) + ((ko >> 4) << 3);
    return ((size_t)(mt * Kt + kt)) * 512 + (size_t)lane * 16 + h;
}
// B matrix (32x16 bf16): lane l<16 holds N=l, halves 0..7 => K=0..7,
// halves 8..15 => K=16..23; lane l>=16 holds N=l-16, K=8..15 / K=24..31.
__device__ __forceinline__ size_t b_frag_idx(int n, int k, int Kt) {
    int nt = n >> 4, kt = k >> 5;
    int no = n & 15, ko = k & 31;
    int lane = no + (((ko >> 3) & 1) << 4);
    int h    = (ko & 7) + ((ko >> 4) << 3);
    return ((size_t)(nt * Kt + kt)) * 512 + (size_t)lane * 16 + h;
}

__device__ __forceinline__ void store_tile(float* __restrict__ C, v8f acc,
                                           int mt, int nt, int lane,
                                           const float* __restrict__ bias,
                                           long long crow,
                                           const int* __restrict__ dlen, int t) {
    // C/D layout: VGPR r, lane<16 -> (M=r, N=lane); lane>=16 -> (M=r+8, N=lane-16)
    int n     = nt * 16 + (lane & 15);
    int mbase = mt * 16 + ((lane >> 4) << 3);
    float bv = bias ? bias[n] : 0.0f;
#pragma unroll
    for (int r = 0; r < 8; ++r) {
        int m = mbase + r;
        float v = acc[r] + bv;
        if (dlen && t >= dlen[m]) v = 0.0f;   // preds * mf
        C[(size_t)m * crow + n] = v;
    }
}

// ---------------------------------------------------------------------------
// 1x1-tile WMMA GEMM (for tiny latency-bound GEMMs in the recurrence)
// ---------------------------------------------------------------------------
__global__ void gemm_bf16_wmma(const unsigned short* __restrict__ Af,
                               const unsigned short* __restrict__ Bf,
                               const float* __restrict__ bias,
                               float* __restrict__ C,
                               int Mt, int Nt, int Kt, long long crow,
                               const int* __restrict__ dlen, int t) {
    int wave = blockIdx.x * (blockDim.x >> 5) + (threadIdx.x >> 5);
    int lane = threadIdx.x & 31;
    if (wave >= Mt * Nt) return;
    int nt = wave / Mt;
    int mt = wave - nt * Mt;
    const v16bf* Ap = (const v16bf*)(Af + (size_t)mt * Kt * 512);
    const v16bf* Bp = (const v16bf*)(Bf + (size_t)nt * Kt * 512);
    v8f acc = {0.f, 0.f, 0.f, 0.f, 0.f, 0.f, 0.f, 0.f};
    for (int kt = 0; kt < Kt; ++kt) {
        v16bf a = Ap[kt * 32 + lane];
        v16bf b = Bp[kt * 32 + lane];
        acc = __builtin_amdgcn_wmma_f32_16x16x32_bf16(
            false, a, false, b, (short)0, acc, false, false);
    }
    store_tile(C, acc, mt, nt, lane, bias, crow, dlen, t);
}

// ---------------------------------------------------------------------------
// 2x2-tile register-blocked WMMA GEMM: 4 fragment loads -> 4 wmma per K-step
// (1 b128 load per wmma).  Odd Mt/Nt handled by clamping (benign duplicate
// recompute+store of identical values).
// ---------------------------------------------------------------------------
__global__ void gemm_bf16_wmma2x2(const unsigned short* __restrict__ Af,
                                  const unsigned short* __restrict__ Bf,
                                  const float* __restrict__ bias,
                                  float* __restrict__ C,
                                  int Mt, int Nt, int Kt, long long crow,
                                  const int* __restrict__ dlen, int t) {
    int wave = blockIdx.x * (blockDim.x >> 5) + (threadIdx.x >> 5);
    int lane = threadIdx.x & 31;
    int Mb = (Mt + 1) >> 1, Nb = (Nt + 1) >> 1;
    if (wave >= Mb * Nb) return;
    int nb = wave / Mb;
    int mb = wave - nb * Mb;
    int mt0 = mb * 2, mt1 = mt0 + 1 < Mt ? mt0 + 1 : Mt - 1;
    int nt0 = nb * 2, nt1 = nt0 + 1 < Nt ? nt0 + 1 : Nt - 1;
    const v16bf* A0 = (const v16bf*)(Af + (size_t)mt0 * Kt * 512);
    const v16bf* A1 = (const v16bf*)(Af + (size_t)mt1 * Kt * 512);
    const v16bf* B0 = (const v16bf*)(Bf + (size_t)nt0 * Kt * 512);
    const v16bf* B1 = (const v16bf*)(Bf + (size_t)nt1 * Kt * 512);
    v8f acc00 = {0.f, 0.f, 0.f, 0.f, 0.f, 0.f, 0.f, 0.f};
    v8f acc01 = acc00, acc10 = acc00, acc11 = acc00;
    for (int kt = 0; kt < Kt; ++kt) {
        int fo = kt * 32 + lane;
        v16bf a0 = A0[fo];
        v16bf a1 = A1[fo];
        v16bf b0 = B0[fo];
        v16bf b1 = B1[fo];
        acc00 = __builtin_amdgcn_wmma_f32_16x16x32_bf16(
            false, a0, false, b0, (short)0, acc00, false, false);
        acc01 = __builtin_amdgcn_wmma_f32_16x16x32_bf16(
            false, a0, false, b1, (short)0, acc01, false, false);
        acc10 = __builtin_amdgcn_wmma_f32_16x16x32_bf16(
            false, a1, false, b0, (short)0, acc10, false, false);
        acc11 = __builtin_amdgcn_wmma_f32_16x16x32_bf16(
            false, a1, false, b1, (short)0, acc11, false, false);
    }
    store_tile(C, acc00, mt0, nt0, lane, bias, crow, dlen, t);
    store_tile(C, acc01, mt0, nt1, lane, bias, crow, dlen, t);
    store_tile(C, acc10, mt1, nt0, lane, bias, crow, dlen, t);
    store_tile(C, acc11, mt1, nt1, lane, bias, crow, dlen, t);
}

// --- sort batch by caption length (stable argsort of -len), B=64 -----------
__global__ void order_kernel(const int* __restrict__ clen,
                             int* __restrict__ order, int* __restrict__ dlen,
                             float* __restrict__ out_order,
                             float* __restrict__ out_dlen) {
    int i = threadIdx.x;
    if (i >= Bc) return;
    int li = clen[i];
    int rank = 0;
    for (int j = 0; j < Bc; ++j) {
        int lj = clen[j];
        if (lj > li || (lj == li && j < i)) rank++;
    }
    order[rank]     = i;
    dlen[rank]      = li - 1;
    out_order[rank] = (float)i;
    out_dlen[rank]  = (float)(li - 1);
}

__global__ void caps_kernel(const int* __restrict__ caps,
                            const int* __restrict__ order,
                            int* __restrict__ caps_s,
                            float* __restrict__ out_caps) {
    int idx = blockIdx.x * blockDim.x + threadIdx.x;
    if (idx >= Bc * Lc) return;
    int b = idx / Lc, l = idx - b * Lc;
    int v = caps[order[b] * Lc + l];
    caps_s[idx]  = v;
    out_caps[idx] = (float)v;
}

// --- weight packing (f32 row-major (N,K) -> bf16 B-fragments) --------------
__global__ void pack_b_generic(const float* __restrict__ W,
                               unsigned short* __restrict__ dst, int N, int K) {
    int idx = blockIdx.x * blockDim.x + threadIdx.x;
    if (idx >= N * K) return;
    int n = idx / K, k = idx - n * K;
    dst[b_frag_idx(n, k, K >> 5)] = f2bf(W[idx]);
}
// fused [W_ih (2048,2560) | W_hh (2048,512)] -> (2048,3072)
__global__ void pack_wcat(const float* __restrict__ Wih,
                          const float* __restrict__ Whh,
                          unsigned short* __restrict__ dst) {
    int idx = blockIdx.x * blockDim.x + threadIdx.x;
    if (idx >= 2048 * KXc) return;
    int n = idx / KXc, k = idx - n * KXc;
    float v = (k < EMBc + Ec) ? Wih[n * (EMBc + Ec) + k]
                              : Whh[n * Dc + (k - EMBc - Ec)];
    dst[b_frag_idx(n, k, KXc >> 5)] = f2bf(v);
}
__global__ void bias_cat_kernel(const float* __restrict__ bih,
                                const float* __restrict__ bhh,
                                float* __restrict__ bc) {
    int i = blockIdx.x * blockDim.x + threadIdx.x;
    if (i < 2048) bc[i] = bih[i] + bhh[i];
}

// --- activation packing (f32 row-major (M,K) -> bf16 A-fragments) ----------
__global__ void pack_a_generic(const float* __restrict__ src,
                               unsigned short* __restrict__ dst, int M, int K) {
    int idx = blockIdx.x * blockDim.x + threadIdx.x;
    if (idx >= M * K) return;
    int m = idx / K, k = idx - m * K;
    dst[a_frag_idx(m, k, K >> 5)] = f2bf(src[idx]);
}
// enc rows reordered by `order` while packing (M = B*P, K = E)
__global__ void pack_enc(const float* __restrict__ enc0,
                         const int* __restrict__ order,
                         unsigned short* __restrict__ dst) {
    long long idx = (long long)blockIdx.x * blockDim.x + threadIdx.x;
    if (idx >= (long long)Bc * Pc * Ec) return;
    int m = (int)(idx / Ec), k = (int)(idx - (long long)m * Ec);
    int b = m / Pc, p = m - b * Pc;
    float v = enc0[((size_t)order[b] * Pc + p) * Ec + k];
    dst[a_frag_idx(m, k, Ec >> 5)] = f2bf(v);
}
// x = [emb_t | gated awe | h]  (M=64, K=3072) straight into A-fragments
__global__ void pack_x(const int* __restrict__ caps_s,
                       const float* __restrict__ emb,
                       const float* __restrict__ awe_g,
                       const float* __restrict__ h,
                       unsigned short* __restrict__ dst, int t) {
    int idx = blockIdx.x * blockDim.x + threadIdx.x;
    if (idx >= Bc * KXc) return;
    int m = idx / KXc, k = idx - m * KXc;
    float v;
    if (k < EMBc)            v = emb[(size_t)caps_s[m * Lc + t] * EMBc + k];
    else if (k < EMBc + Ec)  v = awe_g[m * Ec + (k - EMBc)];
    else                     v = h[m * Dc + (k - EMBc - Ec)];
    dst[a_frag_idx(m, k, KXc >> 5)] = f2bf(v);
}

// --- mean over pixels ------------------------------------------------------
__global__ void mean_enc_kernel(const float* __restrict__ enc0,
                                const int* __restrict__ order,
                                float* __restrict__ mean_enc) {
    int idx = blockIdx.x * blockDim.x + threadIdx.x;
    if (idx >= Bc * Ec) return;
    int b = idx >> 11, c = idx & (Ec - 1);
    const float* row = enc0 + (size_t)order[b] * Pc * Ec + c;
    float s = 0.f;
    for (int p = 0; p < Pc; ++p) s += row[(size_t)p * Ec];
    mean_enc[idx] = s * (1.0f / Pc);
}

// --- attention: e = relu(att1+att2)·w_fa, softmax_p, awe = enc·alpha, gate --
__global__ void attention_kernel(const float* __restrict__ att1,
                                 const float* __restrict__ att2,
                                 const float* __restrict__ w_fa,
                                 const float* __restrict__ b_fa,
                                 const float* __restrict__ enc0,
                                 const int* __restrict__ order,
                                 const float* __restrict__ gate_lin,
                                 float* __restrict__ awe_g,
                                 const int* __restrict__ dlen, int t,
                                 float* __restrict__ alpha_out) {
    __shared__ float s_att2[Ac];
    __shared__ float s_e[Pc];
    __shared__ float s_alpha[Pc];
    __shared__ float s_red[256];
    int b = blockIdx.x, tid = threadIdx.x;
    for (int a = tid; a < Ac; a += 256) s_att2[a] = att2[b * Ac + a];
    __syncthreads();
    int wv = tid >> 5, ln = tid & 31;
    for (int p = wv; p < Pc; p += 8) {            // 8 waves over 196 pixels
        const float* row = att1 + ((size_t)(b * Pc + p)) * Ac;
        if (p + 8 < Pc)                            // global_prefetch_b8 next row
            __builtin_prefetch(row + 8 * Ac + ln, 0, 3);
        float s = 0.f;
        for (int a = ln; a < Ac; a += 32) {
            float v = row[a] + s_att2[a];
            v = v > 0.f ? v : 0.f;
            s += v * w_fa[a];
        }
        for (int off = 16; off; off >>= 1) s += __shfl_xor(s, off, 32);
        if (ln == 0) s_e[p] = s + b_fa[0];
    }
    __syncthreads();
    // softmax over P=196
    s_red[tid] = (tid < Pc) ? s_e[tid] : -1e30f;
    __syncthreads();
    for (int st = 128; st; st >>= 1) {
        if (tid < st) s_red[tid] = fmaxf(s_red[tid], s_red[tid + st]);
        __syncthreads();
    }
    float mx = s_red[0];
    __syncthreads();
    float ev = (tid < Pc) ? __expf(s_e[tid] - mx) : 0.f;
    s_red[tid] = ev;
    __syncthreads();
    for (int st = 128; st; st >>= 1) {
        if (tid < st) s_red[tid] += s_red[tid + st];
        __syncthreads();
    }
    float inv = 1.0f / s_red[0];
    float mf = (t < dlen[b]) ? 1.f : 0.f;
    if (tid < Pc) {
        float al = ev * inv;
        s_alpha[tid] = al;
        alpha_out[((size_t)b * Tc + t) * Pc + tid] = al * mf;
    }
    __syncthreads();
    // awe[c] = sum_p enc[b,p,c] * alpha[p]; then gate (L2-resident traffic)
    const float* erow = enc0 + (size_t)order[b] * Pc * Ec;
    for (int c = tid; c < Ec; c += 256) {
        float acc = 0.f;
        for (int p = 0; p < Pc; ++p) acc += erow[(size_t)p * Ec + c] * s_alpha[p];
        float g = sigmoidf_(gate_lin[b * Ec + c]);
        awe_g[b * Ec + c] = g * acc;
    }
}

// --- LSTM cell elementwise + masked state update + repack h/h2 fragments ---
__global__ void lstm_kernel(const float* __restrict__ gates,
                            float* __restrict__ h, float* __restrict__ c,
                            unsigned short* __restrict__ hfrag,
                            unsigned short* __restrict__ h2frag,
                            const int* __restrict__ dlen, int t) {
    int idx = blockIdx.x * blockDim.x + threadIdx.x;
    if (idx >= Bc * Dc) return;
    int b = idx >> 9, j = idx & (Dc - 1);
    const float* g = gates + (size_t)b * (4 * Dc);
    float i_ = sigmoidf_(g[j]);
    float f_ = sigmoidf_(g[Dc + j]);
    float g_ = tanhf(g[2 * Dc + j]);
    float o_ = sigmoidf_(g[3 * Dc + j]);
    float c2 = f_ * c[idx] + i_ * g_;
    float h2 = o_ * tanhf(c2);
    bool mask = t < dlen[b];
    float hn = mask ? h2 : h[idx];
    float cn = mask ? c2 : c[idx];
    h[idx] = hn;
    c[idx] = cn;
    size_t fi = a_frag_idx(b, j, Dc >> 5);
    hfrag[fi]  = f2bf(hn);   // masked h for next step
    h2frag[fi] = f2bf(h2);   // unmasked h2 for vocab projection
}

// ---------------------------------------------------------------------------
extern "C" void kernel_launch(void* const* d_in, const int* in_sizes, int n_in,
                              void* d_out, int out_size, void* d_ws, size_t ws_size,
                              hipStream_t stream) {
    const float* enc0   = (const float*)d_in[0];
    const int*   caps   = (const int*)d_in[1];
    const int*   clen   = (const int*)d_in[2];
    const float* W_ea   = (const float*)d_in[3];
    const float* b_ea   = (const float*)d_in[4];
    const float* W_da   = (const float*)d_in[5];
    const float* b_da   = (const float*)d_in[6];
    const float* w_fa   = (const float*)d_in[7];
    const float* b_fa   = (const float*)d_in[8];
    const float* emb    = (const float*)d_in[9];
    const float* W_ih   = (const float*)d_in[10];
    const float* b_ih   = (const float*)d_in[11];
    const float* W_hh   = (const float*)d_in[12];
    const float* b_hh   = (const float*)d_in[13];
    const float* W_init_h = (const float*)d_in[14];
    const float* b_init_h = (const float*)d_in[15];
    const float* W_init_c = (const float*)d_in[16];
    const float* b_init_c = (const float*)d_in[17];
    const float* W_fb   = (const float*)d_in[18];
    const float* b_fb   = (const float*)d_in[19];
    const float* W_fc   = (const float*)d_in[20];
    const float* b_fc   = (const float*)d_in[21];

    // ---- output layout (floats, concatenated in return order) ----
    float* out_pred  = (float*)d_out;                       // (B,T,V)
    float* out_caps  = out_pred + (size_t)Bc * Tc * Vc;     // (B,L)
    float* out_dlen  = out_caps + (size_t)Bc * Lc;          // (B,)
    float* out_alpha = out_dlen + Bc;                       // (B,T,P)
    float* out_order = out_alpha + (size_t)Bc * Tc * Pc;    // (B,)

    // ---- workspace carve-out ----
    char* ws = (char*)d_ws;
    size_t off = 0;
    auto carve = [&](size_t bytes) -> char* {
        char* p = ws + off;
        off = (off + bytes + 255) & ~(size_t)255;
        return p;
    };
    int*   order_i  = (int*)carve(Bc * 4);
    int*   dlen_i   = (int*)carve(Bc * 4);
    int*   caps_s   = (int*)carve((size_t)Bc * Lc * 4);
    float* bias_c   = (float*)carve(2048 * 4);
    unsigned short* Wea_f  = (unsigned short*)carve((size_t)Ac * Ec * 2);
    unsigned short* Wda_f  = (unsigned short*)carve((size_t)Ac * Dc * 2);
    unsigned short* Wcat_f = (unsigned short*)carve((size_t)2048 * KXc * 2);
    unsigned short* Wfb_f  = (unsigned short*)carve((size_t)Ec * Dc * 2);
    unsigned short* Wfc_f  = (unsigned short*)carve((size_t)Vc * Dc * 2);
    unsigned short* Wih_f  = (unsigned short*)carve((size_t)Dc * Ec * 2); // W_init_h
    unsigned short* Wic_f  = (unsigned short*)carve((size_t)Dc * Ec * 2); // W_init_c
    float* mean_e   = (float*)carve((size_t)Bc * Ec * 4);
    unsigned short* me_frag  = (unsigned short*)carve((size_t)Bc * Ec * 2);
    unsigned short* enc_frag = (unsigned short*)carve((size_t)Bc * Pc * Ec * 2);
    float* att1     = (float*)carve((size_t)Bc * Pc * Ac * 4);
    float* hbuf     = (float*)carve((size_t)Bc * Dc * 4);
    float* cbuf     = (float*)carve((size_t)Bc * Dc * 4);
    unsigned short* hfrag  = (unsigned short*)carve((size_t)Bc * Dc * 2);
    unsigned short* h2frag = (unsigned short*)carve((size_t)Bc * Dc * 2);
    float* att2     = (float*)carve((size_t)Bc * Ac * 4);
    float* gate_lin = (float*)carve((size_t)Bc * Ec * 4);
    float* awe_g    = (float*)carve((size_t)Bc * Ec * 4);
    float* gatesb   = (float*)carve((size_t)Bc * 4 * Dc * 4);
    unsigned short* xfrag = (unsigned short*)carve((size_t)Bc * KXc * 2);
    (void)ws_size; (void)n_in; (void)in_sizes; (void)out_size;

    auto blk = [](long long n) { return (unsigned)((n + 255) / 256); };
    // pick 1x1 kernel for tiny latency-bound GEMMs, 2x2 blocked for the rest
    auto gemm = [&](const unsigned short* A, const unsigned short* B,
                    const float* bias, float* C, int Mt, int Nt, int Kt,
                    long long crow, const int* dl, int t) {
        if (Mt * Nt <= 128) {
            gemm_bf16_wmma<<<(Mt * Nt + 7) / 8, 256, 0, stream>>>(
                A, B, bias, C, Mt, Nt, Kt, crow, dl, t);
        } else {
            int tiles = ((Mt + 1) >> 1) * ((Nt + 1) >> 1);
            gemm_bf16_wmma2x2<<<(tiles + 7) / 8, 256, 0, stream>>>(
                A, B, bias, C, Mt, Nt, Kt, crow, dl, t);
        }
    };

    // ---- setup: sort, caps gather, weight packing ----
    order_kernel<<<1, 64, 0, stream>>>(clen, order_i, dlen_i, out_order, out_dlen);
    caps_kernel<<<blk(Bc * Lc), 256, 0, stream>>>(caps, order_i, caps_s, out_caps);
    pack_b_generic<<<blk((long long)Ac * Ec), 256, 0, stream>>>(W_ea, Wea_f, Ac, Ec);
    pack_b_generic<<<blk((long long)Ac * Dc), 256, 0, stream>>>(W_da, Wda_f, Ac, Dc);
    pack_wcat<<<blk((long long)2048 * KXc), 256, 0, stream>>>(W_ih, W_hh, Wcat_f);
    pack_b_generic<<<blk((long long)Ec * Dc), 256, 0, stream>>>(W_fb, Wfb_f, Ec, Dc);
    pack_b_generic<<<blk((long long)Vc * Dc), 256, 0, stream>>>(W_fc, Wfc_f, Vc, Dc);
    pack_b_generic<<<blk((long long)Dc * Ec), 256, 0, stream>>>(W_init_h, Wih_f, Dc, Ec);
    pack_b_generic<<<blk((long long)Dc * Ec), 256, 0, stream>>>(W_init_c, Wic_f, Dc, Ec);
    bias_cat_kernel<<<blk(2048), 256, 0, stream>>>(b_ih, b_hh, bias_c);

    // ---- init state: mean_enc -> h0, c0 ----
    mean_enc_kernel<<<blk((long long)Bc * Ec), 256, 0, stream>>>(enc0, order_i, mean_e);
    pack_a_generic<<<blk((long long)Bc * Ec), 256, 0, stream>>>(mean_e, me_frag, Bc, Ec);
    gemm(me_frag, Wih_f, b_init_h, hbuf, Bc / 16, Dc / 16, Ec / 32, Dc, nullptr, 0);
    gemm(me_frag, Wic_f, b_init_c, cbuf, Bc / 16, Dc / 16, Ec / 32, Dc, nullptr, 0);
    pack_a_generic<<<blk((long long)Bc * Dc), 256, 0, stream>>>(hbuf, hfrag, Bc, Dc);

    // ---- one-time: att1 = enc_sorted @ W_ea^T + b_ea  (12544x512, K=2048) ----
    pack_enc<<<blk((long long)Bc * Pc * Ec), 256, 0, stream>>>(enc0, order_i, enc_frag);
    gemm(enc_frag, Wea_f, b_ea, att1, (Bc * Pc) / 16, Ac / 16, Ec / 32, Ac, nullptr, 0);

    // ---- sequential decode: 51 steps ----
    for (int t = 0; t < Tc; ++t) {
        // att2 = h @ W_da^T + b_da        (64x512, K=512)
        gemm(hfrag, Wda_f, b_da, att2, Bc / 16, Ac / 16, Dc / 32, Ac, nullptr, 0);
        // gate_lin = h @ W_fb^T + b_fb    (64x2048, K=512)
        gemm(hfrag, Wfb_f, b_fb, gate_lin, Bc / 16, Ec / 16, Dc / 32, Ec, nullptr, 0);
        // softmax attention + gated context
        attention_kernel<<<Bc, 256, 0, stream>>>(att1, att2, w_fa, b_fa, enc0,
                                                 order_i, gate_lin, awe_g,
                                                 dlen_i, t, out_alpha);
        // x = [emb_t | awe_g | h] packed to A-fragments
        pack_x<<<blk((long long)Bc * KXc), 256, 0, stream>>>(caps_s, emb, awe_g,
                                                             hbuf, xfrag, t);
        // gates = x @ [W_ih|W_hh]^T + (b_ih + b_hh)   (64x2048, K=3072)
        gemm(xfrag, Wcat_f, bias_c, gatesb, Bc / 16, (4 * Dc) / 16, KXc / 32,
             4 * Dc, nullptr, 0);
        // LSTM cell + masked state update, repack h/h2 fragments
        lstm_kernel<<<blk((long long)Bc * Dc), 256, 0, stream>>>(
            gatesb, hbuf, cbuf, hfrag, h2frag, dlen_i, t);
        // preds = h2 @ W_fc^T + b_fc, masked, into (B,T,V) slice t
        gemm(h2frag, Wfc_f, b_fc, out_pred + (size_t)t * Vc,
             Bc / 16, Vc / 16, Dc / 32, (long long)Tc * Vc, dlen_i, t);
    }
}
